// Attention_19550691131655
// MI455X (gfx1250) — matrix operand order repaired and checked by reading, not compiled, not run on hardware
//
#include <hip/hip_runtime.h>
#include <hip/hip_bf16.h>

typedef __attribute__((ext_vector_type(16))) _Float16 v16h;
typedef __attribute__((ext_vector_type(8)))  float    v8f;

#define BATCH 16
#define NTOK  16385          // N = 128*128 + 1
#define TOK   16384          // tokens (rows 1..N-1)
#define CDIM  64
#define LN_EPSF 1e-5f

// ---------------- ws layout ----------------
// s_pos  : BATCH*64 f32
// s_neg  : BATCH*64 f32
// counts : BATCH    u32
// Vp     : BATCH*3*64 f32          (v @ Wp, per batch)
// Ml     : BATCH*64*16 f16         (M[c][j] = 0.125*sum_d Wq[c,d]*k[j,d]; cols 3..15 zero)

__global__ void attn_zero_ws(float* ws) {
    for (int i = threadIdx.x; i < BATCH * CDIM * 2 + BATCH; i += blockDim.x)
        ws[i] = 0.0f;
}

// pass1: mask_weight / mask / mask_prob + masked token sums
__global__ void attn_pass1(const float* __restrict__ x,
                           float* __restrict__ mask_out,
                           float* __restrict__ prob_out,
                           float* __restrict__ s_pos,
                           float* __restrict__ s_neg,
                           unsigned* __restrict__ counts) {
    __shared__ float swt[CDIM];
    __shared__ float spos[CDIM];
    __shared__ float sneg[CDIM];
    __shared__ unsigned scnt;

    const int tid = threadIdx.x;
    const int b   = blockIdx.x >> 6;              // 64 blocks per batch
    const int t   = ((blockIdx.x & 63) << 8) + tid;

    if (tid < CDIM) {
        swt[tid]  = x[((size_t)b * NTOK) * CDIM + tid];  // wt row
        spos[tid] = 0.0f;
        sneg[tid] = 0.0f;
    }
    if (tid == 0) scnt = 0u;
    __syncthreads();

    const float4* xr4 = (const float4*)(x + ((size_t)b * NTOK + 1 + t) * CDIM);
    float4 xv[16];
#pragma unroll
    for (int i = 0; i < 16; ++i) xv[i] = xr4[i];

    float mw = 0.0f;
#pragma unroll
    for (int i = 0; i < 16; ++i) {
        mw += xv[i].x * swt[4*i+0] + xv[i].y * swt[4*i+1]
            + xv[i].z * swt[4*i+2] + xv[i].w * swt[4*i+3];
    }

    const float prob = 1.0f / (1.0f + __expf(-mw));
    const bool  bit  = (mw > 0.0f);               // sigmoid(mw) > 0.5
    prob_out[(size_t)b * TOK + t] = prob;
    mask_out[(size_t)b * TOK + t] = bit ? 1.0f : 0.0f;

    if (bit) atomicAdd(&scnt, 1u);
    float* acc = bit ? spos : sneg;
#pragma unroll
    for (int i = 0; i < 16; ++i) {
        atomicAdd(&acc[4*i+0], mw * xv[i].x);
        atomicAdd(&acc[4*i+1], mw * xv[i].y);
        atomicAdd(&acc[4*i+2], mw * xv[i].z);
        atomicAdd(&acc[4*i+3], mw * xv[i].w);
    }
    __syncthreads();

    if (tid < CDIM)            atomicAdd(&s_pos[b * CDIM + tid], spos[tid]);
    else if (tid < 2 * CDIM)   atomicAdd(&s_neg[b * CDIM + tid - CDIM], sneg[tid - CDIM]);
    if (tid == 0 && scnt)      atomicAdd(&counts[b], scnt);
}

// mid: per-batch fixup + LayerNorm + kv + fold Wq/Wp into tiny M (f16) and Vp
__global__ void attn_mid(const float* __restrict__ x,
                         const float* __restrict__ Wq,
                         const float* __restrict__ Wkv,
                         const float* __restrict__ ln_g,
                         const float* __restrict__ ln_b,
                         const float* __restrict__ Wp,
                         const float* __restrict__ s_pos,
                         const float* __restrict__ s_neg,
                         const unsigned* __restrict__ counts,
                         float* __restrict__ Vp,
                         _Float16* __restrict__ Ml) {
    const int b = blockIdx.x;
    const int c = threadIdx.x;                    // 64 threads

    __shared__ float red[CDIM];
    __shared__ float xsh[3][CDIM];
    __shared__ float kk[3][CDIM];
    __shared__ float vv[3][CDIM];

    auto blocksum = [&](float v) -> float {
        red[c] = v; __syncthreads();
        for (int s = 32; s > 0; s >>= 1) {
            if (c < s) red[c] += red[c + s];
            __syncthreads();
        }
        float r = red[0]; __syncthreads();
        return r;
    };

    const float wt = x[((size_t)b * NTOK) * CDIM + c];
    const float sp = s_pos[b * CDIM + c];
    const float sn = s_neg[b * CDIM + c];
    const bool  any = counts[b] > 0u;
    float rows[3];
    rows[0] = wt;
    rows[1] = any ? sp : (sp + sn);               // mask falls back to all-ones
    rows[2] = any ? sn : 0.0f;

#pragma unroll
    for (int j = 0; j < 3; ++j) {
        const float mu  = blocksum(rows[j]) * (1.0f / CDIM);
        const float dcv = rows[j] - mu;
        const float var = blocksum(dcv * dcv) * (1.0f / CDIM);
        xsh[j][c] = dcv * rsqrtf(var + LN_EPSF) * ln_g[c] + ln_b[c];
    }
    __syncthreads();

#pragma unroll
    for (int j = 0; j < 3; ++j) {
        float ks = 0.0f, vs = 0.0f;
        for (int d = 0; d < CDIM; ++d) {
            const float xvl = xsh[j][d];
            ks += xvl * Wkv[d * 2 * CDIM + c];
            vs += xvl * Wkv[d * 2 * CDIM + CDIM + c];
        }
        kk[j][c] = ks;
        vv[j][c] = vs;
    }
    __syncthreads();

    // M[c][j] = scale * sum_d Wq[c,d] * k[j,d]   (scale = 64^-0.5 = 0.125)
    // stored padded to 16 columns so pass2's B loads are unconditional
#pragma unroll
    for (int j = 0; j < 3; ++j) {
        float m = 0.0f;
        for (int d = 0; d < CDIM; ++d) m += Wq[c * CDIM + d] * kk[j][d];
        Ml[((size_t)b * CDIM + c) * 16 + j] = (_Float16)(0.125f * m);
    }
#pragma unroll
    for (int j = 3; j < 16; ++j)
        Ml[((size_t)b * CDIM + c) * 16 + j] = (_Float16)0.0f;

    // Vp[j][c] = sum_d v[j,d] * Wp[d,c]
#pragma unroll
    for (int j = 0; j < 3; ++j) {
        float s = 0.0f;
        for (int d = 0; d < CDIM; ++d) s += vv[j][d] * Wp[d * CDIM + c];
        Vp[((size_t)b * 3 + j) * CDIM + c] = s;
    }
}

// fixup: if a batch had zero masked tokens, mask becomes all ones
__global__ void attn_fixup_mask(const unsigned* __restrict__ counts,
                                float* __restrict__ mask_out) {
    const int b = blockIdx.x;
    if (counts[b] != 0u) return;
    for (int t = threadIdx.x; t < TOK; t += blockDim.x)
        mask_out[(size_t)b * TOK + t] = 1.0f;
}

// pass2: fused  logits = x@M (WMMA f16)  ->  softmax(3)  ->  out = attn@Vp + bp
__global__ void __launch_bounds__(128)
attn_pass2(const float* __restrict__ x,
           const float* __restrict__ bp,
           const float* __restrict__ Vp,
           const _Float16* __restrict__ Ml,
           float* __restrict__ outp) {
    const int wave = threadIdx.x >> 5;
    const int lane = threadIdx.x & 31;
    const int n16  = lane & 15;
    const int hi   = lane >> 4;
    const int b    = blockIdx.y;

    const int tile = blockIdx.x * 4 + wave;       // 16-row tiles, 1025 per batch
    if (tile >= (NTOK + 15) / 16) return;
    const int r0 = tile * 16;

    // per-lane contiguous float4 slice of Vp rows and bias (cols 4*n16..4*n16+3)
    const float4 vp0 = ((const float4*)(Vp + ((size_t)b * 3 + 0) * CDIM))[n16];
    const float4 vp1 = ((const float4*)(Vp + ((size_t)b * 3 + 1) * CDIM))[n16];
    const float4 vp2 = ((const float4*)(Vp + ((size_t)b * 3 + 2) * CDIM))[n16];
    const float4 bpv = ((const float4*)bp)[n16];

    // B fragment: M is 64x16 (K x N) padded, cols >= 3 are zero — unconditional loads
    const _Float16* Mb = Ml + (size_t)b * CDIM * 16;
    v16h B0, B1;
#pragma unroll
    for (int i = 0; i < 16; ++i) {
        const int k = (hi << 3) + (i & 7) + ((i >> 3) << 4);
        B0[i] = Mb[k * 16 + n16];
        B1[i] = Mb[(k + 32) * 16 + n16];
    }

    // A fragment: 16 rows of x, f32 -> f16, CDNA5 16-bit A 16x32 layout.
    // Row index clamped (garbage only lands in C rows that are never stored).
    int row_a = r0 + n16;
    row_a = (row_a < NTOK) ? row_a : (NTOK - 1);
    const float4* xr4 = (const float4*)(x + ((size_t)b * NTOK + row_a) * CDIM);
    float4 p[8];
#pragma unroll
    for (int s = 0; s < 4; ++s) {                 // K segments 8*hi + {0,16,32,48}
        p[2*s]   = xr4[4*s + 2*hi];
        p[2*s+1] = xr4[4*s + 2*hi + 1];
    }
    const float* pf = (const float*)p;            // pf[0..31] per-lane K slice
    v16h A0, A1;
#pragma unroll
    for (int i = 0; i < 16; ++i) {
        A0[i] = (_Float16)pf[i];                  // K = 8*hi+(i&7) (+16 for i>=8)
        A1[i] = (_Float16)pf[16 + i];             // same + 32
    }

    v8f c = {};
    c = __builtin_amdgcn_wmma_f32_16x16x32_f16(false, A0, false, B0, (short)0, c, false, false);
    c = __builtin_amdgcn_wmma_f32_16x16x32_f16(false, A1, false, B1, (short)0, c, false, false);

    // softmax over the 3 valid logit columns, then out = attn @ Vp + bp
    const int sel = (lane & 16) << 2;             // ds_bpermute byte index base
#pragma unroll
    for (int j = 0; j < 8; ++j) {
        const float lv = c[j];
        const float l0 = __int_as_float(__builtin_amdgcn_ds_bpermute(sel + 0, __float_as_int(lv)));
        const float l1 = __int_as_float(__builtin_amdgcn_ds_bpermute(sel + 4, __float_as_int(lv)));
        const float l2 = __int_as_float(__builtin_amdgcn_ds_bpermute(sel + 8, __float_as_int(lv)));
        const float m  = fmaxf(l0, fmaxf(l1, l2));
        const float e0 = __expf(l0 - m);
        const float e1 = __expf(l1 - m);
        const float e2 = __expf(l2 - m);
        const float inv = __builtin_amdgcn_rcpf(e0 + e1 + e2);
        const float a0 = e0 * inv, a1 = e1 * inv, a2 = e2 * inv;

        const int row = r0 + j + (hi << 3);
        if (row < NTOK) {
            float4 o;
            o.x = a0 * vp0.x + a1 * vp1.x + a2 * vp2.x + bpv.x;
            o.y = a0 * vp0.y + a1 * vp1.y + a2 * vp2.y + bpv.y;
            o.z = a0 * vp0.z + a1 * vp1.z + a2 * vp2.z + bpv.z;
            o.w = a0 * vp0.w + a1 * vp1.w + a2 * vp2.w + bpv.w;
            ((float4*)(outp + ((size_t)b * NTOK + row) * CDIM))[n16] = o;
        }
    }
}

extern "C" void kernel_launch(void* const* d_in, const int* in_sizes, int n_in,
                              void* d_out, int out_size, void* d_ws, size_t ws_size,
                              hipStream_t stream) {
    const float* x    = (const float*)d_in[0];
    const float* Wq   = (const float*)d_in[1];
    const float* Wkv  = (const float*)d_in[2];
    const float* ln_g = (const float*)d_in[3];
    const float* ln_b = (const float*)d_in[4];
    const float* Wp   = (const float*)d_in[5];
    const float* bp   = (const float*)d_in[6];

    float* out      = (float*)d_out;                                // (B,N,C)
    float* mask_out = out + (size_t)BATCH * NTOK * CDIM;            // (B,T,1)
    float* prob_out = mask_out + (size_t)BATCH * TOK;               // (B,T,1)

    float*    s_pos  = (float*)d_ws;
    float*    s_neg  = s_pos + BATCH * CDIM;
    unsigned* counts = (unsigned*)(s_neg + BATCH * CDIM);
    float*    Vp     = (float*)(counts + BATCH);
    _Float16* Ml     = (_Float16*)(Vp + BATCH * 3 * CDIM);

    attn_zero_ws<<<1, 256, 0, stream>>>((float*)d_ws);
    attn_pass1<<<BATCH * (TOK / 256), 256, 0, stream>>>(x, mask_out, prob_out,
                                                        s_pos, s_neg, counts);
    attn_mid<<<BATCH, 64, 0, stream>>>(x, Wq, Wkv, ln_g, ln_b, Wp,
                                       s_pos, s_neg, counts, Vp, Ml);
    attn_fixup_mask<<<BATCH, 256, 0, stream>>>(counts, mask_out);

    const int tiles = (NTOK + 15) / 16;                   // 1025
    dim3 grid2((tiles + 3) / 4, BATCH);
    attn_pass2<<<grid2, 128, 0, stream>>>(x, bp, Vp, Ml, out);
}
